// LSTM_68745246539793
// MI455X (gfx1250) — matrix operand order repaired and checked by reading, not compiled, not run on hardware
//
#include <hip/hip_runtime.h>
#include <hip/hip_bf16.h>
#include <stdint.h>

// ---------------------------------------------------------------------------
// LSTM for MI455X (gfx1250): bf16 WMMA, async global->LDS staging (ASYNCcnt),
// register-resident recurrent weights, persistent kernel with one grid
// barrier per timestep.   T=512, B=64, I=512, H=512, gates = 4H = 2048
// ---------------------------------------------------------------------------

#define T_STEPS 512
#define BATCH   64
#define IN_DIM  512
#define HID     512
#define GATES   2048              // 4*HID
#define KTOT    1024              // IN_DIM + HID
#define MROWS   32768             // T_STEPS * BATCH

typedef __bf16 bf16_t;
typedef __attribute__((ext_vector_type(16))) __bf16 v16bf;
typedef __attribute__((ext_vector_type(8)))  __bf16 v8bf;
typedef __attribute__((ext_vector_type(8)))  float  v8f;

// ---- CDNA5 async global->LDS copy (tracked by ASYNCcnt) -------------------
// GLOBAL_LOAD_ASYNC_TO_LDS_B128: vdst = LDS byte offset (wave-relative),
// vaddr = 64-bit global address, GV mode (saddr = off).
__device__ inline void async_copy_b128(unsigned lds_off, uint64_t gaddr) {
  asm volatile("global_load_async_to_lds_b128 %0, %1, off"
               :: "v"(lds_off), "v"(gaddr)
               : "memory");
}
__device__ inline void wait_asynccnt0() {
  asm volatile("s_wait_asynccnt 0" ::: "memory");
}
// Low 32 bits of a generic LDS pointer == wave-relative LDS byte offset.
__device__ inline unsigned lds_offset_of(const void* p) {
  return (unsigned)(uintptr_t)p;
}

// A-operand loader for V_WMMA_F32_16X16X32_BF16.
// Lane holds row M=lane%16; elements 0..15 hold contiguous
// K = base0..base0+15 (base0 = lane<16?0:16), matching the B operand.
__device__ inline v16bf load_a16x32(const bf16_t* row, int k0) {
  v8bf lo = *(const v8bf*)(row + k0);
  v8bf hi = *(const v8bf*)(row + k0 + 8);
  return __builtin_shufflevector(lo, hi, 0,1,2,3,4,5,6,7,8,9,10,11,12,13,14,15);
}

__device__ inline v8f wmma_bf16(v16bf a, v16bf b, v8f c) {
  // (neg_a, A, neg_b, B, c_mod, C, reuse_a, reuse_b)
  return __builtin_amdgcn_wmma_f32_16x16x32_bf16(false, a, false, b, (short)0, c,
                                                 false, false);
}

__device__ inline float fast_sigmoid(float x) {
  return 1.0f / (1.0f + __expf(-x));
}

// ---------------------------------------------------------------------------
// Kernel 1: prep.  Wt[c][k] = bf16(W[k][c]) (column-major bf16 weights so the
// WMMA B operand is a contiguous 32B load), x -> bf16, init h/c state.
// ---------------------------------------------------------------------------
__global__ __launch_bounds__(256)
void lstm_prep_kernel(const float* __restrict__ x, const float* __restrict__ h0,
                      const float* __restrict__ c0, const float* __restrict__ W,
                      bf16_t* __restrict__ Wt, bf16_t* __restrict__ x_bf,
                      bf16_t* __restrict__ h_bf, float* __restrict__ c_st) {
  long i = (long)blockIdx.x * blockDim.x + threadIdx.x;
  long stride = (long)gridDim.x * blockDim.x;
  for (long idx = i; idx < (long)MROWS * IN_DIM; idx += stride) {
    x_bf[idx] = (bf16_t)x[idx];
    if (idx < (long)GATES * KTOT) {                 // transpose W -> Wt
      long c = idx >> 10;                           // / KTOT
      long k = idx & (KTOT - 1);
      Wt[idx] = (bf16_t)W[k * GATES + c];
    }
    if (idx < (long)BATCH * HID) {
      h_bf[idx] = (bf16_t)h0[idx];
      c_st[idx] = c0[idx];
    }
  }
}

// ---------------------------------------------------------------------------
// Kernel 2: gates_x = X @ W_x + b  (M=32768, N=2048, K=512).
// 8 waves/block share one 16x512 A tile: async-staged into LDS once, then
// each wave sweeps K with 16 chained v_wmma (A from ds_load, B from global).
// ---------------------------------------------------------------------------
__global__ __launch_bounds__(256)
void lstm_xgemm_kernel(const bf16_t* __restrict__ x_bf, const bf16_t* __restrict__ Wt,
                       const float* __restrict__ bias, float* __restrict__ gx) {
  __shared__ bf16_t As[16 * IN_DIM];                // 16 KB A tile

  const int lane = threadIdx.x & 31;
  const int warp = threadIdx.x >> 5;
  const int wid  = blockIdx.x * 8 + warp;           // 0 .. 262143
  const int n_t  = wid & 127;                       // N tile (2048/16)
  const int m_t  = wid >> 7;                        // M tile; same for all 8 waves
  const int col  = n_t * 16 + (lane & 15);
  const int hi8  = (lane & 16) ? 8 : 0;

  // Async-stage the block's A tile (rows m_t*16 .. +15, contiguous 16 KB).
  {
    const unsigned lds_base = lds_offset_of(As);
    const uint64_t gbase = (uint64_t)(uintptr_t)(x_bf + (size_t)m_t * 16 * IN_DIM);
#pragma unroll
    for (int j = 0; j < 4; ++j) {
      const unsigned q = (unsigned)threadIdx.x + j * 256;   // 1024 x 16B chunks
      async_copy_b128(lds_base + q * 16, gbase + (uint64_t)q * 16);
    }
    wait_asynccnt0();
  }
  __syncthreads();

  const bf16_t* arow = As + (lane & 15) * IN_DIM;
  // B layout: lane holds column 'col'; per half-wave K offset 0/16, contiguous.
  const bf16_t* bcol = Wt + (long)col * KTOT + (hi8 << 1);   // x-part rows [0,512)

  v8f acc = {};
#pragma unroll
  for (int kc = 0; kc < 16; ++kc) {
    v16bf a = load_a16x32(arow, kc * 32 + (hi8 << 1)); // ds_load_b128 x2
    v16bf b = *(const v16bf*)(bcol + kc * 32);      // global_load_b128 x2
    acc = wmma_bf16(a, b, acc);
  }

  const float bv = bias[col];
  float* out = gx + (long)(m_t * 16) * GATES + col;
#pragma unroll
  for (int v = 0; v < 8; ++v) out[(long)(v + hi8) * GATES] = acc[v] + bv;
}

// ---------------------------------------------------------------------------
// Kernel 3: persistent recurrent kernel. 64 blocks x 8 waves = 512 waves;
// wave (block m,jp ; warp g,jsub) owns gate tile n_t = g*32 + jp*2 + jsub and
// keeps its 16-column W_h slice (16KB) in 128 VGPRs/lane for all 512 steps.
// Per step: grid barrier -> async-stage the block's 16x512 h slab into LDS ->
// 16 WMMAs (B resident in VGPRs) -> LDS gate combine -> cell update.
// ---------------------------------------------------------------------------
#define NBLK 64

__device__ inline void grid_barrier(unsigned* __restrict__ counter, int gen) {
  __threadfence();                                   // publish this thread's stores
  __syncthreads();
  if (threadIdx.x == 0) {
    __hip_atomic_fetch_add(counter, 1u, __ATOMIC_RELEASE, __HIP_MEMORY_SCOPE_AGENT);
    const unsigned target = (unsigned)NBLK * (unsigned)gen;
    while (__hip_atomic_load(counter, __ATOMIC_ACQUIRE, __HIP_MEMORY_SCOPE_AGENT)
           < target) {
      __builtin_amdgcn_s_sleep(2);
    }
  }
  __syncthreads();
}

__global__ __launch_bounds__(256)
void lstm_recurrent_kernel(const bf16_t* __restrict__ Wt, const float* __restrict__ gx,
                           bf16_t* __restrict__ h_bf, float* __restrict__ c_st,
                           float* __restrict__ out, unsigned* __restrict__ counter) {
  __shared__ float  gsm[8 * 256];                    // 8 gate tiles, 16x16 f32
  __shared__ bf16_t As[16 * HID];                    // 16 KB h slab for this m

  const int lane = threadIdx.x & 31;
  const int warp = threadIdx.x >> 5;
  const int m    = blockIdx.x & 3;                   // batch tile (64/16)
  const int jp   = blockIdx.x >> 2;                  // hidden 32-col group (0..15)
  const int g    = warp & 3;                         // gate: 0=f 1=i 2=g 3=o
  const int jsub = warp >> 2;                        // 0..1
  const int n_t  = g * 32 + jp * 2 + jsub;           // gate-matrix column tile
  const int col  = n_t * 16 + (lane & 15);
  const int hi8  = (lane & 16) ? 8 : 0;

  // Preload this wave's W_h slice into registers (K rows 512..1023 of Wt).
  const bf16_t* bcol = Wt + (long)col * KTOT + IN_DIM + (hi8 << 1);
  v16bf breg[16];
#pragma unroll
  for (int kc = 0; kc < 16; ++kc) breg[kc] = *(const v16bf*)(bcol + kc * 32);

  const unsigned lds_base = lds_offset_of(As);
  const uint64_t hbase = (uint64_t)(uintptr_t)(h_bf + (size_t)m * 16 * HID);
  const bf16_t* arow = As + (lane & 15) * HID;

  for (int t = 0; t < T_STEPS; ++t) {
    if (t) grid_barrier(counter, t);                 // h_{t-1} from all blocks visible

    // Async-stage h rows m*16..+15 (contiguous 16 KB) into LDS.
#pragma unroll
    for (int j = 0; j < 4; ++j) {
      const unsigned q = (unsigned)threadIdx.x + j * 256;
      async_copy_b128(lds_base + q * 16, hbase + (uint64_t)q * 16);
    }

    const float* gxt = gx + ((long)t * BATCH + m * 16) * GATES + col;
    __builtin_prefetch(gxt, 0, 1);                   // global_prefetch_b8

    wait_asynccnt0();
    __syncthreads();                                 // all waves' As chunks landed

    // gates = h_{t-1} @ W_h  (16 chained WMMAs, B resident in VGPRs)
    v8f acc = {};
#pragma unroll
    for (int kc = 0; kc < 16; ++kc) {
      v16bf a = load_a16x32(arow, kc * 32 + (hi8 << 1)); // ds_load_b128 x2
      acc = wmma_bf16(a, breg[kc], acc);
    }
    // += precomputed x-part (already includes bias)
#pragma unroll
    for (int v = 0; v < 8; ++v) acc[v] += gxt[(long)(v + hi8) * GATES];

    // Stash this wave's 16x16 tile in LDS for the gate combine.
    float* sm = gsm + warp * 256 + (lane & 15);
#pragma unroll
    for (int v = 0; v < 8; ++v) sm[(v + hi8) * 16] = acc[v];
    __syncthreads();

    // Elementwise LSTM cell update over the block's 16x32 (batch x hidden) patch.
#pragma unroll
    for (int ee = 0; ee < 2; ++ee) {
      const int e   = threadIdx.x + ee * 256;        // 0..511
      const int r   = e >> 5;                        // batch row in tile
      const int hc  = e & 31;                        // hidden col in block
      const int js  = hc >> 4;
      const int c16 = hc & 15;
      const int base = r * 16 + c16;
      const float fg = gsm[(js * 4 + 0) * 256 + base];
      const float ig = gsm[(js * 4 + 1) * 256 + base];
      const float gg = gsm[(js * 4 + 2) * 256 + base];
      const float og = gsm[(js * 4 + 3) * 256 + base];
      const float f  = fast_sigmoid(fg);
      const float i  = fast_sigmoid(ig);
      const float cd = tanhf(gg);
      const float o  = fast_sigmoid(og);
      const long gi = (long)(m * 16 + r) * HID + (jp * 32 + hc);
      const float cn = f * c_st[gi] + i * cd;
      c_st[gi] = cn;
      const float hn = o * tanhf(cn);
      out[(long)t * (BATCH * HID) + gi] = hn;        // output sequence h_t
      h_bf[gi] = (bf16_t)hn;                         // feed next step's WMMA
      if (t == T_STEPS - 1) {
        out[(long)T_STEPS * BATCH * HID + gi] = hn;                // h_T
        out[(long)T_STEPS * BATCH * HID + BATCH * HID + gi] = cn;  // c_T
      }
    }
    // next-iteration grid_barrier() provides the intra-block sync before LDS reuse
  }
}

// ---------------------------------------------------------------------------
// Host launcher
// ---------------------------------------------------------------------------
extern "C" void kernel_launch(void* const* d_in, const int* in_sizes, int n_in,
                              void* d_out, int out_size, void* d_ws, size_t ws_size,
                              hipStream_t stream) {
  const float* x    = (const float*)d_in[0];   // [T,B,I]
  const float* h0   = (const float*)d_in[1];   // [1,B,H]
  const float* c0   = (const float*)d_in[2];   // [1,B,H]
  const float* W    = (const float*)d_in[3];   // [I+H, 4H]
  const float* bias = (const float*)d_in[4];   // [4H]
  float* out = (float*)d_out;

  char* ws = (char*)d_ws;
  size_t off = 0;
  unsigned* counter = (unsigned*)(ws + off); off += 256;
  float*   c_st = (float*)(ws + off);  off += (size_t)BATCH * HID * 4;       // 128 KB
  bf16_t*  h_bf = (bf16_t*)(ws + off); off += (size_t)BATCH * HID * 2;       //  64 KB
  bf16_t*  Wt   = (bf16_t*)(ws + off); off += (size_t)GATES * KTOT * 2;      //   4 MB
  bf16_t*  x_bf = (bf16_t*)(ws + off); off += (size_t)MROWS * IN_DIM * 2;    //  32 MB
  float*   gx   = (float*)(ws + off);  off += (size_t)MROWS * GATES * 4;     // 256 MB

  hipMemsetAsync(counter, 0, 256, stream);

  lstm_prep_kernel<<<65536, 256, 0, stream>>>(x, h0, c0, W, Wt, x_bf, h_bf, c_st);
  lstm_xgemm_kernel<<<32768, 256, 0, stream>>>(x_bf, Wt, bias, gx);
  lstm_recurrent_kernel<<<NBLK, 256, 0, stream>>>(Wt, gx, h_bf, c_st, out, counter);
}
